// GraphAttention_1898375545033
// MI455X (gfx1250) — matrix operand order repaired and checked by reading, not compiled, not run on hardware
//
#include <hip/hip_runtime.h>

typedef __attribute__((ext_vector_type(16))) __bf16 v16bf;
typedef __attribute__((ext_vector_type(8)))  __bf16 v8bf;
typedef __attribute__((ext_vector_type(8)))  float  v8f;

#define GAT_N 2048
#define GAT_B 8
#define GAT_C 128
#define GAT_D 64
#define GAT_KH 1024   // K processed per LDS stage (2 stages of 1024)
#define GAT_PAD 8     // bf16 elements of row padding -> stride 1032 halves = 516 dwords, 516%64=4 -> conflict-free

// ---------------------------------------------------------------------------
// Kernel 1: hidden = x @ W (fp32), transposed bf16 copy HbT[b][d][n] for WMMA-B,
//           s1[b,n] = hidden . a[:D], s2[b,n] = hidden . a[D:]
// One block per (b,n); 64 threads (one per d).
// ---------------------------------------------------------------------------
__global__ void __launch_bounds__(64)
gat_hidden_kernel(const float* __restrict__ x, const float* __restrict__ W,
                  const float* __restrict__ a,
                  float* __restrict__ hidden, __bf16* __restrict__ HbT,
                  float* __restrict__ s1, float* __restrict__ s2)
{
    const int bn = blockIdx.x;               // 0 .. B*N-1
    const int b  = bn >> 11;
    const int n  = bn & (GAT_N - 1);
    const int d  = threadIdx.x;              // 0..63

    const float* xrow = x + (size_t)bn * GAT_C;
    float acc = 0.0f;
#pragma unroll 4
    for (int k = 0; k < GAT_C; ++k)
        acc = fmaf(xrow[k], W[k * GAT_D + d], acc);

    hidden[(size_t)bn * GAT_D + d] = acc;
    HbT[((size_t)b * GAT_D + d) * GAT_N + n] = (__bf16)acc;

    __shared__ float sh[2][GAT_D];
    sh[0][d] = acc * a[d];
    sh[1][d] = acc * a[GAT_D + d];
    __syncthreads();
    if (d == 0) {                            // deterministic sequential reduce
        float t1 = 0.0f, t2 = 0.0f;
        for (int k = 0; k < GAT_D; ++k) { t1 += sh[0][k]; t2 += sh[1][k]; }
        s1[bn] = t1;
        s2[bn] = t2;
    }
}

// Faithful reproduction of the reference's cat/view index quirk.
__device__ __forceinline__ float gat_logit(const float* __restrict__ s1b,
                                           const float* __restrict__ s2b,
                                           int i, int j)
{
    const int NN = GAT_N * GAT_N;
    int m  = i * GAT_N + j;
    int r1 = 2 * m, r2 = 2 * m + 1;
    int idx1 = (r1 < NN) ? (r1 >> 11) : (r1 & (GAT_N - 1));
    int idx2 = (r2 < NN) ? (r2 >> 11) : (r2 & (GAT_N - 1));
    return s1b[idx1] + s2b[idx2];
}

// ---------------------------------------------------------------------------
// Kernel 2: fused masked-leakyrelu-softmax + bf16 WMMA (att @ hidden)
//           + bias + tanh + residual.
// Block = (i-tile of 16 rows, batch), 128 threads = 4 waves.
// Softmax: 8 lanes per row (shuffle reductions only).
// Matmul: wave w owns output cols [16w,16w+16); K staged in two 1024 halves.
// ---------------------------------------------------------------------------
__global__ void __launch_bounds__(128)
gat_attn_kernel(const int* __restrict__ adj,
                const float* __restrict__ hidden,
                const __bf16* __restrict__ HbT,
                const float* __restrict__ s1, const float* __restrict__ s2,
                const float* __restrict__ bias,
                float* __restrict__ out)
{
    const int it  = blockIdx.x;              // 0..127
    const int b   = blockIdx.y;              // 0..7
    const int i0  = it * 16;
    const int tid = threadIdx.x;             // 0..127
    const int lane = tid & 31;
    const int wv   = tid >> 5;               // wave id 0..3

    __shared__ __bf16 Pb[16][GAT_KH + GAT_PAD];
    __shared__ float  rinv[16];

    // --- softmax-phase mapping: 8 threads per row ---
    const int r   = tid >> 3;                // 0..15  (row within tile)
    const int sub = tid & 7;                 // 0..7
    const int i   = i0 + r;
    const float* s1b = s1 + (size_t)b * GAT_N;
    const float* s2b = s2 + (size_t)b * GAT_N;
    const int*   adjrow = adj + (size_t)i * GAT_N;

    // Pass 1: full-row max of masked leaky-relu logits
    float mx = -3.0e38f;
    for (int j = sub; j < GAT_N; j += 8) {
        float e = gat_logit(s1b, s2b, i, j);
        e = (e > 0.0f) ? e : 0.2f * e;
        e = (adjrow[j] > 0) ? e : -9.0e15f;
        mx = fmaxf(mx, e);
    }
    mx = fmaxf(mx, __shfl_xor(mx, 1));
    mx = fmaxf(mx, __shfl_xor(mx, 2));
    mx = fmaxf(mx, __shfl_xor(mx, 4));       // 8-lane group stays inside wave32

    // --- WMMA fragment mapping (wave32) ---
    const int ncol0 = wv * 16;
    const int arow  = lane & 15;             // A: M = lane (both halves)
    const int asel  = lane >> 4;             // A: K sub-block select
    const int bcol  = lane & 15;             // B: N = lane
    const int bsel  = lane >> 4;             // B: K half select
    const __bf16* Bbase =
        HbT + ((size_t)b * GAT_D + (ncol0 + bcol)) * (size_t)GAT_N;

    v8f acc = {};
    float psum = 0.0f;

#pragma unroll
    for (int half = 0; half < 2; ++half) {
        const int jbase = half * GAT_KH;

        // Pass 2: exp(e - max) -> bf16 probs in LDS (unnormalized), track sum
        for (int j = sub; j < GAT_KH; j += 8) {
            const int jj = jbase + j;
            float e = gat_logit(s1b, s2b, i, jj);
            e = (e > 0.0f) ? e : 0.2f * e;
            e = (adjrow[jj] > 0) ? e : -9.0e15f;
            float p = __expf(e - mx);
            psum += p;
            Pb[r][j] = (__bf16)p;
        }
        __syncthreads();

        // WMMA over this K half: 32 x v_wmma_f32_16x16x32_bf16 per wave
        for (int k0 = 0; k0 < GAT_KH; k0 += 32) {
            // A fragment: contiguous K order used by this code (matches B)
            //   lanes 0-15 : K = k0+{0..15}
            //   lanes 16-31: K = k0+16+{0..15}
            const v8bf alo = *(const v8bf*)&Pb[arow][k0 + asel * 16];
            const v8bf ahi = *(const v8bf*)&Pb[arow][k0 + asel * 16 + 8];
            v16bf afrag;
#pragma unroll
            for (int t = 0; t < 8; ++t) { afrag[t] = alo[t]; afrag[8 + t] = ahi[t]; }

            // B fragment: same K order as A -> per lane 16 contiguous K
            //   lanes 0-15: K = k0+0..15, lanes 16-31: K = k0+16..31
            const v16bf bfrag =
                *(const v16bf*)(Bbase + jbase + k0 + bsel * 16);

            acc = __builtin_amdgcn_wmma_f32_16x16x32_bf16(
                      false, afrag, false, bfrag, (short)0, acc, false, false);
        }
        __syncthreads();
    }

    // Row sums -> 1/sum
    psum += __shfl_xor(psum, 1);
    psum += __shfl_xor(psum, 2);
    psum += __shfl_xor(psum, 4);
    if (sub == 0) rinv[r] = 1.0f / psum;
    __syncthreads();

    // D fragment (f32 16x16): VGPR t, lanes 0-15: M=t; lanes 16-31: M=t+8
#pragma unroll
    for (int t = 0; t < 8; ++t) {
        const int M   = t + 8 * (lane >> 4);
        const int row = i0 + M;
        const int col = ncol0 + (lane & 15);
        const size_t o = ((size_t)b * GAT_N + row) * GAT_D + col;
        float v = acc[t] * rinv[M];
        out[o] = tanhf(v + bias[col]) + hidden[o];
    }
}

extern "C" void kernel_launch(void* const* d_in, const int* in_sizes, int n_in,
                              void* d_out, int out_size, void* d_ws, size_t ws_size,
                              hipStream_t stream)
{
    (void)in_sizes; (void)n_in; (void)out_size; (void)ws_size;

    const float* x    = (const float*)d_in[0];   // [8,2048,128]
    const int*   adj  = (const int*)  d_in[1];   // [1,2048,2048]
    const float* W    = (const float*)d_in[2];   // [128,64]
    const float* a    = (const float*)d_in[3];   // [128,1]
    const float* bias = (const float*)d_in[4];   // [64]
    float* out = (float*)d_out;                  // [8,2048,64]

    char* ws = (char*)d_ws;
    float*  hidden = (float*)ws;                                   // 4 MB fp32
    __bf16* HbT    = (__bf16*)(ws + (size_t)GAT_B * GAT_N * GAT_D * 4);  // 2 MB bf16 (transposed)
    float*  s1     = (float*)(ws + (size_t)GAT_B * GAT_N * GAT_D * 6);   // 64 KB
    float*  s2     = s1 + GAT_B * GAT_N;                                 // 64 KB

    gat_hidden_kernel<<<GAT_B * GAT_N, 64, 0, stream>>>(x, W, a, hidden, HbT, s1, s2);
    gat_attn_kernel<<<dim3(GAT_N / 16, GAT_B), 128, 0, stream>>>(
        adj, hidden, HbT, s1, s2, bias, out);
}
